// TSPTWFeasGIN_83416854823073
// MI455X (gfx1250) — matrix operand order repaired and checked by reading, not compiled, not run on hardware
//
#include <hip/hip_runtime.h>
#include <hip/hip_bf16.h>

#define NN   100000
#define NE   1600000
#define HIDD 128
#define NTILES (NN / 16)   // 6250 exactly

typedef __attribute__((ext_vector_type(16))) __bf16 v16bf;
typedef __attribute__((ext_vector_type(8)))  float  v8f;

union Frag16 { v16bf v; uint4 q[2]; };

// ---------------------------------------------------------------------------
// h0 = x @ node_in_w + node_in_b   ([N,4]@[4,128])
// ---------------------------------------------------------------------------
__global__ void proj_kernel(const float* __restrict__ x,
                            const float* __restrict__ w,   // [4][128]
                            const float* __restrict__ b,   // [128]
                            float* __restrict__ h) {
    int gid = blockIdx.x * blockDim.x + threadIdx.x;
    if (gid >= NN * HIDD) return;
    int n = gid >> 7;
    int c = gid & 127;
    float4 xv = *(const float4*)(x + n * 4);
    float acc = b[c];
    acc += xv.x * w[0 * HIDD + c];
    acc += xv.y * w[1 * HIDD + c];
    acc += xv.z * w[2 * HIDD + c];
    acc += xv.w * w[3 * HIDD + c];
    h[gid] = acc;
}

// ---------------------------------------------------------------------------
// zB = hA   (float4 copy; zB then receives the edge scatter-adds)
// ---------------------------------------------------------------------------
__global__ void copy_kernel(const float* __restrict__ src, float* __restrict__ dst) {
    int gid = blockIdx.x * blockDim.x + threadIdx.x;
    if (gid >= NN * HIDD / 4) return;
    ((float4*)dst)[gid] = ((const float4*)src)[gid];
}

// ---------------------------------------------------------------------------
// Edge scatter-add: zB[dst] += hA[src].  One wave (32 lanes) per edge; each
// lane gathers a float4 and issues 4 global_atomic_add_f32.
// ---------------------------------------------------------------------------
__global__ void scatter_kernel(const int* __restrict__ edge,   // [2][NE]
                               const float* __restrict__ hA,
                               float* __restrict__ zB) {
    int gid  = blockIdx.x * blockDim.x + threadIdx.x;
    int e    = gid >> 5;
    int lane = gid & 31;
    if (e >= NE) return;
    int s = edge[e];
    int d = edge[NE + e];
    float4 v = *(const float4*)(hA + s * HIDD + lane * 4);
    float* p = zB + d * HIDD + lane * 4;
    __hip_atomic_fetch_add(p + 0, v.x, __ATOMIC_RELAXED, __HIP_MEMORY_SCOPE_AGENT);
    __hip_atomic_fetch_add(p + 1, v.y, __ATOMIC_RELAXED, __HIP_MEMORY_SCOPE_AGENT);
    __hip_atomic_fetch_add(p + 2, v.z, __ATOMIC_RELAXED, __HIP_MEMORY_SCOPE_AGENT);
    __hip_atomic_fetch_add(p + 3, v.w, __ATOMIC_RELAXED, __HIP_MEMORY_SCOPE_AGENT);
}

// ---------------------------------------------------------------------------
// fp32 -> bf16 conversion (4 elems/thread)
// ---------------------------------------------------------------------------
__global__ void cvt_bf16_kernel(const float* __restrict__ src, __bf16* __restrict__ dst) {
    int gid = blockIdx.x * blockDim.x + threadIdx.x;
    if (gid >= NN * HIDD / 4) return;
    float4 v = ((const float4*)src)[gid];
    union { unsigned long long u; __bf16 h[4]; } o;
    o.h[0] = (__bf16)v.x; o.h[1] = (__bf16)v.y;
    o.h[2] = (__bf16)v.z; o.h[3] = (__bf16)v.w;
    ((unsigned long long*)dst)[gid] = o.u;
}

// ---------------------------------------------------------------------------
// Repack w1/w2 (fp32 [3][128][128]) into bf16 B-fragment order:
// slot m (0..2 = w1 layers, 3..5 = w2 layers);
// dst[m][ (t*4+c)*32 + lane ][ j ] = W[ c*32 + (lane>>4)*8 + j + (j>=8 ? 8 : 0) ][ t*16 + (lane&15) ]
// -> each wave's B fragment for (t,c) is one contiguous 32B run per lane.
// ---------------------------------------------------------------------------
__global__ void repack_kernel(const float* __restrict__ w1,
                              const float* __restrict__ w2,
                              __bf16* __restrict__ wpk) {
    int gid = blockIdx.x * blockDim.x + threadIdx.x;
    if (gid >= 6 * 1024) return;
    int m    = gid >> 10;
    int r    = gid & 1023;          // (t*4+c)*32 + lane
    int lane = r & 31;
    int tc   = r >> 5;
    int c    = tc & 3;
    int t    = tc >> 2;
    const float* W = (m < 3) ? (w1 + m * HIDD * HIDD) : (w2 + (m - 3) * HIDD * HIDD);
    int n  = t * 16 + (lane & 15);
    int k0 = c * 32 + (lane >> 4) * 8;
    __bf16* dst = wpk + m * (HIDD * HIDD) + r * 16;
#pragma unroll
    for (int j = 0; j < 16; ++j) dst[j] = (__bf16)W[(k0 + j + (j >= 8 ? 8 : 0)) * HIDD + n];
}

// ---------------------------------------------------------------------------
// GEMM: out = relu(A @ W + bias), A: [NN,128] bf16 row-major, W: packed bf16.
// One wave per 16-row tile; 8 N-tiles x 4 K-chunks = 32 v_wmma_f32_16x16x32_bf16.
// BF16_OUT=true writes bf16 (feeds next GEMM), else fp32.
// ---------------------------------------------------------------------------
template <bool BF16_OUT>
__global__ void gin_gemm_kernel(const __bf16* __restrict__ A,
                                const __bf16* __restrict__ Wpk,
                                const float* __restrict__ bias,
                                void* __restrict__ out) {
    int wid  = (blockIdx.x * blockDim.x + threadIdx.x) >> 5;
    int lane = threadIdx.x & 31;
    if (wid >= NTILES) return;                  // wave-uniform: EXEC stays all-1s

    int half = lane >> 4;                       // 16-bit A-fragment layout (ISA 7.12.2)
    int rsub = lane & 15;
    int row  = wid * 16 + rsub;

    Frag16 a[4];
    const __bf16* arow = A + row * HIDD;
#pragma unroll
    for (int c = 0; c < 4; ++c) {
        const __bf16* p = arow + c * 32 + half * 8;
        a[c].q[0] = *(const uint4*)p;           // K = c*32 + half*8 .. +7
        a[c].q[1] = *(const uint4*)(p + 16);    // K = c*32 + 16 + half*8 .. +7
    }

#pragma unroll
    for (int t = 0; t < 8; ++t) {
        v8f acc = {};
#pragma unroll
        for (int c = 0; c < 4; ++c) {
            Frag16 b;
            const __bf16* bp = Wpk + ((t * 4 + c) * 32 + lane) * 16;
            b.q[0] = *(const uint4*)bp;
            b.q[1] = *(const uint4*)(bp + 8);
            acc = __builtin_amdgcn_wmma_f32_16x16x32_bf16(
                false, a[c].v, false, b.v, (short)0, acc, false, false);
        }
        int ncol = t * 16 + rsub;
        float bv = bias[ncol];
#pragma unroll
        for (int r = 0; r < 8; ++r) {           // D: VGPR r, lane -> M = r + 8*half
            float v = acc[r] + bv;
            v = v > 0.f ? v : 0.f;
            int m = wid * 16 + r + 8 * half;
            if (BF16_OUT) ((__bf16*)out)[m * HIDD + ncol] = (__bf16)v;
            else          ((float*)out)[m * HIDD + ncol]  = v;
        }
    }
}

// ---------------------------------------------------------------------------
// logits[n] = dot(h[n], out_w) + out_b.  One wave per node, shuffle reduction.
// ---------------------------------------------------------------------------
__global__ void out_kernel(const float* __restrict__ h,
                           const float* __restrict__ ow,   // [128]
                           const float* __restrict__ ob,   // [1]
                           float* __restrict__ logits) {
    int gid  = blockIdx.x * blockDim.x + threadIdx.x;
    int n    = gid >> 5;
    int lane = gid & 31;
    if (n >= NN) return;
    float4 hv = *(const float4*)(h + n * HIDD + lane * 4);
    float4 wv = *(const float4*)(ow + lane * 4);
    float p = hv.x * wv.x + hv.y * wv.y + hv.z * wv.z + hv.w * wv.w;
#pragma unroll
    for (int off = 16; off >= 1; off >>= 1) p += __shfl_xor(p, off, 32);
    if (lane == 0) logits[n] = p + ob[0];
}

// ---------------------------------------------------------------------------
extern "C" void kernel_launch(void* const* d_in, const int* in_sizes, int n_in,
                              void* d_out, int out_size, void* d_ws, size_t ws_size,
                              hipStream_t stream) {
    const float* x      = (const float*)d_in[0];
    const int*   edge   = (const int*)d_in[1];
    const float* niw    = (const float*)d_in[2];
    const float* nib    = (const float*)d_in[3];
    const float* w1     = (const float*)d_in[4];
    const float* b1     = (const float*)d_in[5];
    const float* w2     = (const float*)d_in[6];
    const float* b2     = (const float*)d_in[7];
    const float* ow     = (const float*)d_in[8];
    const float* ob     = (const float*)d_in[9];
    float*       logits = (float*)d_out;

    // workspace carve-up (~128 MB total)
    float*  hA  = (float*)d_ws;                        // [NN][128] f32
    float*  zB  = hA + (size_t)NN * HIDD;              // [NN][128] f32 (also bf16 alias)
    __bf16* s1  = (__bf16*)(zB + (size_t)NN * HIDD);   // [NN][128] bf16 (GEMM1 input)
    __bf16* wpk = s1 + (size_t)NN * HIDD;              // 6 * 128*128 bf16 packed weights
    __bf16* s2  = (__bf16*)zB;                         // GEMM1 output reuses zB's storage

    const int B = 256;
    repack_kernel<<<(6 * 1024 + B - 1) / B, B, 0, stream>>>(w1, w2, wpk);
    proj_kernel<<<(NN * HIDD + B - 1) / B, B, 0, stream>>>(x, niw, nib, hA);

    for (int l = 0; l < 3; ++l) {
        copy_kernel<<<(NN * HIDD / 4 + B - 1) / B, B, 0, stream>>>(hA, zB);
        scatter_kernel<<<((size_t)NE * 32 + B - 1) / B, B, 0, stream>>>(edge, hA, zB);
        cvt_bf16_kernel<<<(NN * HIDD / 4 + B - 1) / B, B, 0, stream>>>(zB, s1);
        // z' = relu(z @ w1[l] + b1[l])  -> bf16 (zB storage now free, reused as s2)
        gin_gemm_kernel<true><<<(NTILES * 32 + B - 1) / B, B, 0, stream>>>(
            s1, wpk + (size_t)l * HIDD * HIDD, b1 + l * HIDD, (void*)s2);
        // h  = relu(z' @ w2[l] + b2[l]) -> fp32 hA for next layer's aggregation
        gin_gemm_kernel<false><<<(NTILES * 32 + B - 1) / B, B, 0, stream>>>(
            s2, wpk + (size_t)(3 + l) * HIDD * HIDD, b2 + l * HIDD, (void*)hA);
    }

    out_kernel<<<((size_t)NN * 32 + B - 1) / B, B, 0, stream>>>(hA, ow, ob, logits);
}
